// GAT_56977036148745
// MI455X (gfx1250) — matrix operand order repaired and checked by reading, not compiled, not run on hardware
//
#include <hip/hip_runtime.h>
#include <hip/hip_fp16.h>

typedef __attribute__((ext_vector_type(16))) _Float16 v16h;
typedef __attribute__((ext_vector_type(8)))  _Float16 v8h;
typedef __attribute__((ext_vector_type(8)))  float    v8f;

#define N_NODES   16384
#define N_EDGES   524288
#define E_TOT     (N_EDGES + N_NODES)
#define N_GRAPHS  256
#define LAYER_DIM 256
#define POOL_DIM  768

// ---------- helpers ----------
__device__ __forceinline__ unsigned enc_f(float f) {
  unsigned i = __float_as_uint(f);
  return (i & 0x80000000u) ? ~i : (i | 0x80000000u);
}
__device__ __forceinline__ float dec_f(unsigned u) {
  unsigned i = (u & 0x80000000u) ? (u & 0x7FFFFFFFu) : ~u;
  return __uint_as_float(i);
}

// ---------- fp32 -> f16 flat convert ----------
__global__ __launch_bounds__(256) void k_convert_f16(
    const float* __restrict__ src, _Float16* __restrict__ dst, int n) {
  int i = blockIdx.x * blockDim.x + threadIdx.x;
  if (i < n) dst[i] = (_Float16)src[i];
}

// ---------- weight convert + transpose: fp32 [K,N] -> f16 [N,K] ----------
__global__ __launch_bounds__(256) void k_transpose_f16(
    const float* __restrict__ W, _Float16* __restrict__ Wt, int K, int N) {
  int idx = blockIdx.x * blockDim.x + threadIdx.x;
  if (idx >= K * N) return;
  int k = idx / N, n = idx - k * N;
  Wt[(size_t)n * K + k] = (_Float16)W[idx];
}

__global__ __launch_bounds__(256) void k_zerof(float* __restrict__ p, int n) {
  int i = blockIdx.x * blockDim.x + threadIdx.x;
  if (i < n) p[i] = 0.0f;
}

// out[n, :] = bias  (doubles as the accumulator init for the edge scatter)
__global__ __launch_bounds__(256) void k_bias_init(
    float* __restrict__ out, const float* __restrict__ b) {
  int i = blockIdx.x * blockDim.x + threadIdx.x;
  if (i < N_NODES * LAYER_DIM) out[i] = b[i & (LAYER_DIM - 1)];
}

// ---------- WMMA GEMM: C[M,N] = A[M,K](f16) * Bt[N,K](f16)^T (+bias)(relu) ----------
// one wave computes a 16x64 strip (4 accumulators); K mult of 32, M mult 16, N mult 64.
__global__ __launch_bounds__(256) void k_wmma_gemm(
    const _Float16* __restrict__ A, const _Float16* __restrict__ Bt,
    float* __restrict__ C, const float* __restrict__ bias,
    int M, int N, int K, int relu) {
  int wave    = (blockIdx.x * blockDim.x + threadIdx.x) >> 5;
  int nstrips = N >> 6;
  int total   = (M >> 4) * nstrips;
  if (wave >= total) return;                 // wave-uniform: EXEC all-ones for WMMA
  int tm = wave / nstrips, tn = wave - tm * nstrips;
  int lane = threadIdx.x & 31;
  int l15 = lane & 15, hi = lane >> 4;
  int kg = hi * 16;   // A: per-half K offset     (K = kg..kg+15 contiguous)
  int kh = hi * 16;   // B: per-half K offset     (K = kh..kh+15 contiguous)
  const _Float16* arow  = A  + (size_t)(tm * 16 + l15) * K;
  const _Float16* bbase = Bt + (size_t)(tn * 64 + l15) * K;
  v8f acc0 = {}, acc1 = {}, acc2 = {}, acc3 = {};
  for (int k0 = 0; k0 < K; k0 += 32) {
    v8h alo = *(const v8h*)(arow + k0 + kg);
    v8h ahi = *(const v8h*)(arow + k0 + kg + 8);
    v16h a = __builtin_shufflevector(alo, ahi,
        0, 1, 2, 3, 4, 5, 6, 7, 8, 9, 10, 11, 12, 13, 14, 15);
    v16h b0 = *(const v16h*)(bbase + (size_t)0 * 16 * K + k0 + kh);
    v16h b1 = *(const v16h*)(bbase + (size_t)1 * 16 * K + k0 + kh);
    v16h b2 = *(const v16h*)(bbase + (size_t)2 * 16 * K + k0 + kh);
    v16h b3 = *(const v16h*)(bbase + (size_t)3 * 16 * K + k0 + kh);
    acc0 = __builtin_amdgcn_wmma_f32_16x16x32_f16(false, a, false, b0, (short)0, acc0, false, false);
    acc1 = __builtin_amdgcn_wmma_f32_16x16x32_f16(false, a, false, b1, (short)0, acc1, false, false);
    acc2 = __builtin_amdgcn_wmma_f32_16x16x32_f16(false, a, false, b2, (short)0, acc2, false, false);
    acc3 = __builtin_amdgcn_wmma_f32_16x16x32_f16(false, a, false, b3, (short)0, acc3, false, false);
  }
  v8f accs[4] = {acc0, acc1, acc2, acc3};
#pragma unroll
  for (int sub = 0; sub < 4; ++sub) {
    int col = tn * 64 + sub * 16 + l15;
    float bv = bias ? bias[col] : 0.0f;
#pragma unroll
    for (int v = 0; v < 8; ++v) {            // C/D: row = v + 8*hi
      int r = tm * 16 + v + 8 * hi;
      float val = accs[sub][v] + bv;
      if (relu) val = fmaxf(val, 0.0f);
      C[(size_t)r * N + col] = val;
    }
  }
}

// ---------- edge phase ----------
// one wave per edge; lanes 0-15 = head 0, lanes 16-31 = head 1; 8 ch/lane.
__global__ __launch_bounds__(256) void k_edge_logits(
    const int* __restrict__ ei, const float* __restrict__ xl,
    const float* __restrict__ xr, const float* __restrict__ att,
    float* __restrict__ logits, unsigned* __restrict__ mmax) {
  int e = (blockIdx.x * blockDim.x + threadIdx.x) >> 5;
  if (e >= E_TOT) return;
  int src, dst;
  if (e < N_EDGES) { src = ei[e]; dst = ei[N_EDGES + e]; }
  else             { src = dst = e - N_EDGES; }
  int lane = threadIdx.x & 31;
  int head = lane >> 4;
  int off  = head * 128 + (lane & 15) * 8;
  const float4* pl = (const float4*)(xl + (size_t)src * LAYER_DIM + off);
  const float4* pr = (const float4*)(xr + (size_t)dst * LAYER_DIM + off);
  const float4* pa = (const float4*)(att + off);
  float s = 0.0f;
#pragma unroll
  for (int q = 0; q < 2; ++q) {
    float4 l = pl[q], r = pr[q], a = pa[q];
    float vx = l.x + r.x, vy = l.y + r.y, vz = l.z + r.z, vw = l.w + r.w;
    vx = vx > 0.0f ? vx : 0.2f * vx;
    vy = vy > 0.0f ? vy : 0.2f * vy;
    vz = vz > 0.0f ? vz : 0.2f * vz;
    vw = vw > 0.0f ? vw : 0.2f * vw;
    s += a.x * vx + a.y * vy + a.z * vz + a.w * vw;
  }
#pragma unroll
  for (int m = 1; m < 16; m <<= 1) s += __shfl_xor(s, m, 32);
  if ((lane & 15) == 0) {
    logits[(size_t)e * 2 + head] = s;
    atomicMax(&mmax[dst * 2 + head], enc_f(s));
  }
}

__global__ __launch_bounds__(256) void k_edge_softmax(
    const int* __restrict__ ei, const float* __restrict__ logits,
    const unsigned* __restrict__ mmax, float* __restrict__ ev,
    float* __restrict__ denom) {
  int tid = blockIdx.x * blockDim.x + threadIdx.x;
  if (tid >= E_TOT * 2) return;
  int e = tid >> 1, h = tid & 1;
  int dst = (e < N_EDGES) ? ei[N_EDGES + e] : (e - N_EDGES);
  float m = dec_f(mmax[dst * 2 + h]);
  float v = __expf(logits[tid] - m);
  ev[tid] = v;
  atomicAdd(&denom[dst * 2 + h], v);
}

__global__ __launch_bounds__(256) void k_edge_scatter(
    const int* __restrict__ ei, const float* __restrict__ xl,
    const float* __restrict__ ev, const float* __restrict__ denom,
    float* __restrict__ out) {
  int e = (blockIdx.x * blockDim.x + threadIdx.x) >> 5;
  if (e >= E_TOT) return;
  int src, dst;
  if (e < N_EDGES) { src = ei[e]; dst = ei[N_EDGES + e]; }
  else             { src = dst = e - N_EDGES; }
  int lane = threadIdx.x & 31;
  int head = lane >> 4;
  int off  = head * 128 + (lane & 15) * 8;
  float alpha = ev[(size_t)e * 2 + head] / denom[dst * 2 + head];
  const float* pl = xl + (size_t)src * LAYER_DIM + off;
  float* po = out + (size_t)dst * LAYER_DIM + off;
#pragma unroll
  for (int q = 0; q < 8; ++q) atomicAdd(po + q, alpha * pl[q]);
}

// ---------- pooling + FFN tail ----------
__global__ __launch_bounds__(256) void k_pool(
    const float* __restrict__ h, const int* __restrict__ batch,
    float* __restrict__ pooled, int layerOff) {
  int i = blockIdx.x * blockDim.x + threadIdx.x;
  if (i >= N_NODES * LAYER_DIM) return;
  int n = i >> 8, d = i & 255;
  atomicAdd(&pooled[(size_t)batch[n] * POOL_DIM + layerOff + d], h[i]);
}

__global__ __launch_bounds__(256) void k_ffn2(
    const float* __restrict__ hid, const float* __restrict__ fW2,
    const float* __restrict__ fb2, float* __restrict__ out) {
  int g = (blockIdx.x * blockDim.x + threadIdx.x) >> 5;
  if (g >= N_GRAPHS) return;
  int lane = threadIdx.x & 31;
  float s = 0.0f;
  for (int k = lane; k < POOL_DIM; k += 32)
    s += hid[(size_t)g * POOL_DIM + k] * fW2[k];
#pragma unroll
  for (int m = 1; m < 32; m <<= 1) s += __shfl_xor(s, m, 32);
  if (lane == 0) out[g] = s + fb2[0];
}

// ---------- launch ----------
extern "C" void kernel_launch(void* const* d_in, const int* in_sizes, int n_in,
                              void* d_out, int out_size, void* d_ws, size_t ws_size,
                              hipStream_t stream) {
  (void)in_sizes; (void)n_in; (void)out_size; (void)ws_size;
  const float* x     = (const float*)d_in[0];
  const int*   ei    = (const int*)d_in[1];
  const int*   batch = (const int*)d_in[2];
  const float* Wl[3]  = {(const float*)d_in[3], (const float*)d_in[7],  (const float*)d_in[11]};
  const float* Wr[3]  = {(const float*)d_in[4], (const float*)d_in[8],  (const float*)d_in[12]};
  const float* att[3] = {(const float*)d_in[5], (const float*)d_in[9],  (const float*)d_in[13]};
  const float* bb[3]  = {(const float*)d_in[6], (const float*)d_in[10], (const float*)d_in[14]};
  const float* fW1 = (const float*)d_in[15];
  const float* fb1 = (const float*)d_in[16];
  const float* fW2 = (const float*)d_in[17];
  const float* fb2 = (const float*)d_in[18];

  char* base = (char*)d_ws;
  size_t off = 0;
  auto alloc = [&](size_t bytes) -> void* {
    void* p = base + off;
    off = (off + bytes + 255) & ~(size_t)255;
    return p;
  };
  float*    xl     = (float*)alloc((size_t)N_NODES * LAYER_DIM * 4);
  float*    xr     = (float*)alloc((size_t)N_NODES * LAYER_DIM * 4);
  float*    h[3];
  for (int l = 0; l < 3; ++l) h[l] = (float*)alloc((size_t)N_NODES * LAYER_DIM * 4);
  float*    logits = (float*)alloc((size_t)E_TOT * 2 * 4);
  float*    ev     = (float*)alloc((size_t)E_TOT * 2 * 4);
  unsigned* mmax   = (unsigned*)alloc((size_t)N_NODES * 2 * 4);
  float*    denom  = (float*)alloc((size_t)N_NODES * 2 * 4);
  float*    pooled = (float*)alloc((size_t)N_GRAPHS * POOL_DIM * 4);
  float*    hid    = (float*)alloc((size_t)N_GRAPHS * POOL_DIM * 4);
  _Float16* xh     = (_Float16*)alloc((size_t)N_NODES * LAYER_DIM * 2);   // layer input, f16
  _Float16* pooledh= (_Float16*)alloc((size_t)N_GRAPHS * POOL_DIM * 2);
  const int Kin[3] = {128, 256, 256};
  _Float16* Wlt[3]; _Float16* Wrt[3];
  for (int l = 0; l < 3; ++l) {
    Wlt[l] = (_Float16*)alloc((size_t)LAYER_DIM * Kin[l] * 2);
    Wrt[l] = (_Float16*)alloc((size_t)LAYER_DIM * Kin[l] * 2);
  }
  _Float16* fW1t = (_Float16*)alloc((size_t)POOL_DIM * POOL_DIM * 2);

  auto cdiv = [](long long a, long long b) { return (int)((a + b - 1) / b); };

  // weights -> f16 transposed
  for (int l = 0; l < 3; ++l) {
    int kn = Kin[l] * LAYER_DIM;
    k_transpose_f16<<<cdiv(kn, 256), 256, 0, stream>>>(Wl[l], Wlt[l], Kin[l], LAYER_DIM);
    k_transpose_f16<<<cdiv(kn, 256), 256, 0, stream>>>(Wr[l], Wrt[l], Kin[l], LAYER_DIM);
  }
  k_transpose_f16<<<cdiv(POOL_DIM * POOL_DIM, 256), 256, 0, stream>>>(fW1, fW1t, POOL_DIM, POOL_DIM);

  k_zerof<<<cdiv(N_GRAPHS * POOL_DIM, 256), 256, 0, stream>>>(pooled, N_GRAPHS * POOL_DIM);

  const float* cur = x;
  for (int l = 0; l < 3; ++l) {
    int K = Kin[l];
    // layer input -> f16 once (removes per-tile fp32 reload + cvt in GEMM)
    k_convert_f16<<<cdiv((long long)N_NODES * K, 256), 256, 0, stream>>>(cur, xh, N_NODES * K);
    int waves = (N_NODES / 16) * (LAYER_DIM / 64);
    k_wmma_gemm<<<cdiv((long long)waves * 32, 256), 256, 0, stream>>>(
        xh, Wlt[l], xl, nullptr, N_NODES, LAYER_DIM, K, 0);
    k_wmma_gemm<<<cdiv((long long)waves * 32, 256), 256, 0, stream>>>(
        xh, Wrt[l], xr, nullptr, N_NODES, LAYER_DIM, K, 0);
    k_bias_init<<<cdiv((long long)N_NODES * LAYER_DIM, 256), 256, 0, stream>>>(h[l], bb[l]);
    k_zerof<<<cdiv(N_NODES * 2, 256), 256, 0, stream>>>((float*)mmax, N_NODES * 2);
    k_zerof<<<cdiv(N_NODES * 2, 256), 256, 0, stream>>>(denom, N_NODES * 2);
    k_edge_logits<<<cdiv((long long)E_TOT * 32, 256), 256, 0, stream>>>(
        ei, xl, xr, att[l], logits, mmax);
    k_edge_softmax<<<cdiv((long long)E_TOT * 2, 256), 256, 0, stream>>>(
        ei, logits, mmax, ev, denom);
    k_edge_scatter<<<cdiv((long long)E_TOT * 32, 256), 256, 0, stream>>>(
        ei, xl, ev, denom, h[l]);
    cur = h[l];
  }

  for (int l = 0; l < 3; ++l)
    k_pool<<<cdiv((long long)N_NODES * LAYER_DIM, 256), 256, 0, stream>>>(
        h[l], batch, pooled, l * LAYER_DIM);

  k_convert_f16<<<cdiv((long long)N_GRAPHS * POOL_DIM, 256), 256, 0, stream>>>(
      pooled, pooledh, N_GRAPHS * POOL_DIM);
  int waves2 = (N_GRAPHS / 16) * (POOL_DIM / 64);
  k_wmma_gemm<<<cdiv((long long)waves2 * 32, 256), 256, 0, stream>>>(
      pooledh, fW1t, hid, fb1, N_GRAPHS, POOL_DIM, POOL_DIM, 1);
  k_ffn2<<<cdiv((long long)N_GRAPHS * 32, 256), 256, 0, stream>>>(
      hid, fW2, fb2, (float*)d_out);
}
